// BinaryToCost_15212774163234
// MI455X (gfx1250) — matrix operand order repaired and checked by reading, not compiled, not run on hardware
//
#include <hip/hip_runtime.h>
#include <stdint.h>

// QUBO batch cost: out[b] = x_b^T Q x_b, B=1024, N=2048, fp32 in/out.
// Phase 1: convert x -> bf16 (exact), Q -> transposed bf16 hi/lo split (ws).
// Phase 2: WMMA bf16 GEMM, double-buffered LDS staged via async-to-LDS DMA,
//          fused (S .* X) row-reduce epilogue with f32 atomics.

#define B_ROWS 1024
#define N_DIM  2048

typedef __bf16 bf16x8  __attribute__((ext_vector_type(8)));
typedef __bf16 bf16x16 __attribute__((ext_vector_type(16)));
typedef float  f32x8   __attribute__((ext_vector_type(8)));

constexpr int BM  = 128;  // WG tile rows (batch)
constexpr int BN  = 64;   // WG tile cols (Q columns)
constexpr int BK  = 32;   // K per step (one bf16 WMMA K)
constexpr int LDA = 40;   // padded LDS row stride (halves) -> conflict-free b128
constexpr int LDB = 40;

#if __has_builtin(__builtin_amdgcn_global_load_async_to_lds_b128)
#define HAVE_ASYNC_LDS 1
#else
#define HAVE_ASYNC_LDS 0
#endif

#if __has_builtin(__builtin_amdgcn_s_wait_asynccnt)
#define WAIT_ASYNC(n) __builtin_amdgcn_s_wait_asynccnt(n)
#else
#define WAIT_ASYNC(n) asm volatile("s_wait_asynccnt %0" ::"i"(n) : "memory")
#endif

// Builtin parameter types (from clang diagnostic): int4 in as1 / as3.
typedef int v4i __attribute__((vector_size(16)));
typedef __attribute__((address_space(1))) v4i gv4i;
typedef __attribute__((address_space(3))) v4i lv4i;

__device__ __forceinline__ unsigned short f2bf(float f) {
  unsigned u = __float_as_uint(f);
  u += 0x7fffu + ((u >> 16) & 1u);      // round-to-nearest-even
  return (unsigned short)(u >> 16);
}
__device__ __forceinline__ float bf2f(unsigned short h) {
  return __uint_as_float((unsigned)h << 16);
}

// Move one 16B chunk global->LDS. Async DMA path on gfx1250; sync fallback.
__device__ __forceinline__ void stage16(const unsigned short* g, unsigned short* l) {
#if HAVE_ASYNC_LDS
  // as1 global ptr is numerically identical to flat; low 32 bits of a flat LDS
  // pointer are the wave-relative LDS address (aperture strips to addr[31:0]).
  __builtin_amdgcn_global_load_async_to_lds_b128(
      (gv4i*)(uintptr_t)g, (lv4i*)(unsigned)(uintptr_t)l, 0, 0);
#else
  *(uint4*)l = *(const uint4*)g;
#endif
}

// Build a 16-half WMMA fragment from two 16B LDS chunks.
__device__ __forceinline__ bf16x16 ld_frag(const unsigned short* p, int step_halves) {
  bf16x8 a = *(const bf16x8*)p;
  bf16x8 b = *(const bf16x8*)(p + step_halves);
  return __builtin_shufflevector(a, b, 0,1,2,3,4,5,6,7,8,9,10,11,12,13,14,15);
}

// ---------------- Phase 1a: x (binary fp32) -> bf16, exact ----------------
__global__ __launch_bounds__(256)
void cvt_x_kernel(const float* __restrict__ x, unsigned short* __restrict__ xb) {
  size_t i = ((size_t)blockIdx.x * 256 + threadIdx.x) << 2;
  const float4 v = *(const float4*)(x + i);
  uint2 p;
  p.x = (unsigned)f2bf(v.x) | ((unsigned)f2bf(v.y) << 16);
  p.y = (unsigned)f2bf(v.z) | ((unsigned)f2bf(v.w) << 16);
  *(uint2*)(xb + i) = p;
}

// ------- Phase 1b: Q fp32 -> transposed bf16 hi/lo (LDS tile transpose) ----
__global__ __launch_bounds__(256)
void cvt_q_kernel(const float* __restrict__ Q,
                  unsigned short* __restrict__ Qh,
                  unsigned short* __restrict__ Ql) {
  __shared__ unsigned short th[64 * 72];   // stride 72 halves = 144B (16B aligned)
  __shared__ unsigned short tl[64 * 72];
  const int tid = threadIdx.x;
  const int k0  = (blockIdx.x & 31) << 6;
  const int nn0 = (blockIdx.x >> 5) << 6;
  #pragma unroll
  for (int i = 0; i < 4; ++i) {
    int idx = tid + i * 256;              // 1024 float4 chunks of the 64x64 tile
    int r   = idx >> 4;                   // k row
    int c   = (idx & 15) << 2;            // n col base
    const float4 v = *(const float4*)(Q + (size_t)(k0 + r) * N_DIM + nn0 + c);
    float q[4] = {v.x, v.y, v.z, v.w};
    #pragma unroll
    for (int e = 0; e < 4; ++e) {
      unsigned short hi = f2bf(q[e]);
      unsigned short lo = f2bf(q[e] - bf2f(hi));
      th[(c + e) * 72 + r] = hi;
      tl[(c + e) * 72 + r] = lo;
    }
  }
  __syncthreads();
  #pragma unroll
  for (int i = 0; i < 2; ++i) {
    int idx = tid + i * 256;              // 512 8-half output chunks
    int n   = idx >> 3;
    int c   = (idx & 7) << 3;
    uint4 vh = *(const uint4*)&th[n * 72 + c];
    uint4 vl = *(const uint4*)&tl[n * 72 + c];
    *(uint4*)(Qh + (size_t)(nn0 + n) * N_DIM + k0 + c) = vh;
    *(uint4*)(Ql + (size_t)(nn0 + n) * N_DIM + k0 + c) = vl;
  }
}

// ---------------- Phase 2: WMMA GEMM + fused epilogue ----------------------
__global__ __launch_bounds__(256, 2)
void qubo_gemm_wmma(const unsigned short* __restrict__ xb,
                    const unsigned short* __restrict__ Qh,   // [n][k] bf16 hi
                    const unsigned short* __restrict__ Ql,   // [n][k] bf16 lo
                    const float* __restrict__ x,
                    float* __restrict__ out) {
  __shared__ unsigned short sA [2][BM * LDA];
  __shared__ unsigned short sBh[2][BN * LDB];
  __shared__ unsigned short sBl[2][BN * LDB];

  const int tid  = threadIdx.x;
  const int lane = tid & 31;
  const int wave = tid >> 5;
  const int wm   = wave & 3;      // 4 waves along M (32 rows each)
  const int wn   = wave >> 2;     // 2 waves along N (32 cols each)
  const int half = lane >> 4;
  const int l16  = lane & 15;

  const int bm0 = (blockIdx.x & 7)  * BM;
  const int n0  = (blockIdx.x >> 3) * BN;

  const int bn = tid >> 2;                 // B staging: row per thread
  const int bc = (tid & 3) << 3;           // 8-half chunk within row

  auto stage = [&](int buf, int k0) {      // 4 async b128 per thread
    #pragma unroll
    for (int i = 0; i < 2; ++i) {
      int idx = tid + i * 256;             // 512 chunks: A tile 128x32 halves
      int r = idx >> 2, c = (idx & 3) << 3;
      stage16(xb + (size_t)(bm0 + r) * N_DIM + k0 + c, &sA[buf][r * LDA + c]);
    }
    stage16(Qh + (size_t)(n0 + bn) * N_DIM + k0 + bc, &sBh[buf][bn * LDB + bc]);
    stage16(Ql + (size_t)(n0 + bn) * N_DIM + k0 + bc, &sBl[buf][bn * LDB + bc]);
  };

  f32x8 acc[2][2] = {};

  stage(0, 0);
  constexpr int KS = N_DIM / BK;           // 64 K-steps
  for (int s = 0; s < KS; ++s) {
    const int buf = s & 1;
    if (s + 1 < KS) {
      stage(buf ^ 1, (s + 1) * BK);        // tile s+1 in flight during compute
#if HAVE_ASYNC_LDS
      WAIT_ASYNC(4);                       // tile s (first 4 in-order) complete
#endif
    } else {
#if HAVE_ASYNC_LDS
      WAIT_ASYNC(0);
#endif
    }
    __syncthreads();

    // Fragments per ISA wave32 16-bit layouts.
    const int ak = half * 8;               // A: halves {ak..+7} and {ak+16..+23}
    const int bk = half * 8;               // B: halves {bk..+7} and {bk+16..+23}, n-major rows
    bf16x16 af[2], bh[2], bl[2];
    #pragma unroll
    for (int mi = 0; mi < 2; ++mi)
      af[mi] = ld_frag(&sA[buf][(wm * 32 + mi * 16 + l16) * LDA + ak], 16);
    #pragma unroll
    for (int ni = 0; ni < 2; ++ni) {
      bh[ni] = ld_frag(&sBh[buf][(wn * 32 + ni * 16 + l16) * LDB + bk], 16);
      bl[ni] = ld_frag(&sBl[buf][(wn * 32 + ni * 16 + l16) * LDB + bk], 16);
    }

    #pragma unroll
    for (int mi = 0; mi < 2; ++mi)
      #pragma unroll
      for (int ni = 0; ni < 2; ++ni) {
        acc[mi][ni] = __builtin_amdgcn_wmma_f32_16x16x32_bf16(
            false, af[mi], false, bh[ni], (short)0, acc[mi][ni], false, false);
        acc[mi][ni] = __builtin_amdgcn_wmma_f32_16x16x32_bf16(
            false, af[mi], false, bl[ni], (short)0, acc[mi][ni], false, false);
      }
    __syncthreads();
  }

  // Fused epilogue: out[b] += sum_n S[b][n] * x[b][n]
  // C/D layout: VGPR r holds M = 8*half + r, N = l16.
  const int gn0 = n0 + wn * 32 + l16;
  #pragma unroll
  for (int mi = 0; mi < 2; ++mi) {
    #pragma unroll
    for (int r = 0; r < 8; ++r) {
      int gm  = bm0 + wm * 32 + mi * 16 + half * 8 + r;
      float p = acc[mi][0][r] * x[(size_t)gm * N_DIM + gn0]
              + acc[mi][1][r] * x[(size_t)gm * N_DIM + gn0 + 16];
      p += __shfl_xor(p, 1);
      p += __shfl_xor(p, 2);
      p += __shfl_xor(p, 4);
      p += __shfl_xor(p, 8);               // row sum across each 16-lane half
      if (l16 == 0) atomicAdd(out + gm, p);
    }
  }
}

extern "C" void kernel_launch(void* const* d_in, const int* in_sizes, int n_in,
                              void* d_out, int out_size, void* d_ws, size_t ws_size,
                              hipStream_t stream) {
  (void)in_sizes; (void)n_in; (void)ws_size;
  const float* x = (const float*)d_in[0];   // [1024 x 2048] fp32 (binary values)
  const float* Q = (const float*)d_in[1];   // [2048 x 2048] fp32
  float* out = (float*)d_out;               // [1024] fp32

  unsigned short* Qh = (unsigned short*)d_ws;                 // 8 MB
  unsigned short* Ql = Qh + (size_t)N_DIM * N_DIM;            // 8 MB
  unsigned short* xb = Ql + (size_t)N_DIM * N_DIM;            // 4 MB

  (void)hipMemsetAsync(out, 0, (size_t)out_size * sizeof(float), stream);

  cvt_x_kernel<<<dim3((B_ROWS * N_DIM) / (4 * 256)), dim3(256), 0, stream>>>(x, xb);
  cvt_q_kernel<<<dim3((N_DIM / 64) * (N_DIM / 64)), dim3(256), 0, stream>>>(Q, Qh, Ql);

  dim3 grid((B_ROWS / BM) * (N_DIM / BN));  // 8 * 32 = 256 workgroups
  qubo_gemm_wmma<<<grid, dim3(256), 0, stream>>>(xb, Qh, Ql, x, out);
}
